// AttentionWithBias_56762287784061
// MI455X (gfx1250) — matrix operand order repaired and checked by reading, not compiled, not run on hardware
//
#include <hip/hip_runtime.h>
#include <cstdint>
#include <cstddef>

// CDNA5 (gfx1250) WMMA fragment types
typedef __attribute__((ext_vector_type(16))) __bf16       v16bf;
typedef __attribute__((ext_vector_type(8)))  float        v8f;
typedef __attribute__((ext_vector_type(8)))  unsigned int v8u;
// TDM descriptor groups (clang-23 6-arg builtin signature)
typedef __attribute__((ext_vector_type(4)))  unsigned int v4u;
typedef __attribute__((ext_vector_type(8)))  int          v8i;
typedef __attribute__((ext_vector_type(4)))  int          v4i;

constexpr int   B_    = 2;
constexpr int   H_    = 16;
constexpr int   L_    = 2048;
constexpr int   D_    = 64;
constexpr int   BM    = 64;      // query rows per workgroup (4 waves x 16)
constexpr int   BN    = 64;      // kv cols per iteration
constexpr int   KSTR  = 66;      // padded bf16 row stride (33 dwords -> conflict-free)
constexpr int   BSTR  = 66;      // padded f32 dword stride for TDM bias tile
constexpr float SCALE = 0.125f;  // 1/sqrt(64)
constexpr float NEGV  = -999999.0f;

#if defined(__gfx1250__) && __has_builtin(__builtin_amdgcn_tensor_load_to_lds)
#define USE_TDM 1
#else
#define USE_TDM 0
#endif

#if USE_TDM
// Issue a TDM DMA of a 64x64 f32 tile (row stride L_) from global into LDS,
// with LDS padding: every 64 dwords insert 2 dwords -> row stride 66 dwords,
// which makes the per-lane bias reads bank-conflict-free across both
// half-wave row groups (8*66 mod 64 == 16).
__device__ __forceinline__ void tdm_load_bias_tile(const float* gsrc, float* ldst) {
    const uint64_t ga  = (uint64_t)(uintptr_t)gsrc;
    const uint32_t lds = (uint32_t)(uintptr_t)ldst;   // generic->LDS: low 32 bits

    v4u g0;
    g0[0] = 1u;                                        // count=1, user descriptor
    g0[1] = lds;                                       // lds_addr
    g0[2] = (uint32_t)ga;                              // global_addr[31:0]
    g0[3] = (uint32_t)((ga >> 32) & 0x01FFFFFFu)       // global_addr[56:32]
          | (2u << 30);                                // type = 2 ("image")

    v8i g1;
    g1[0] = (int)((2u << 16)        // data_size = 4 bytes
                | (1u << 20)        // pad_enable
                | (5u << 22)        // pad_interval code 5 = 64 dwords
                | (1u << 25));      // pad_amount   code 1 = 2 dwords
    g1[1] = (int)((2048u & 0xFFFFu) << 16);  // tensor_dim0[15:0] (atomic_barrier_addr=0)
    g1[2] = (int)(((2048u >> 16) & 0xFFFFu)  // tensor_dim0[31:16]
                | ((2048u & 0xFFFFu) << 16));// tensor_dim1[15:0]
    g1[3] = (int)(((2048u >> 16) & 0xFFFFu)  // tensor_dim1[31:16]
                | (64u << 16));              // tile_dim0 = 64
    g1[4] = (int)(64u);                      // tile_dim1 = 64, tile_dim2 = 0 (2D)
    g1[5] = (int)(2048u);                    // tensor_dim0_stride[31:0] = L
    g1[6] = 0;                               // stride0[47:32]=0, dim1_stride lo=0
    g1[7] = 0;                               // dim1_stride hi = 0

    v4i z4 = {0, 0, 0, 0};                   // groups 2/3 unused (<=2D tile)
    v8i z8 = {0, 0, 0, 0, 0, 0, 0, 0};       // extra group (clang-23 form), unused
    __builtin_amdgcn_tensor_load_to_lds(g0, g1, z4, z4, z8, 0);
}
#endif

__global__ __launch_bounds__(128, 1)
void attn_bias_fwd_kernel(const float*   __restrict__ Q,
                          const float*   __restrict__ Kp,
                          const float*   __restrict__ Vp,
                          const uint8_t* __restrict__ QM,
                          const uint8_t* __restrict__ VM,
                          const float*   __restrict__ Bias,
                          float*         __restrict__ Out)
{
    __shared__ __bf16 sK [BN * KSTR];        // K tile, row = kv, col = d
    __shared__ __bf16 sVT[D_ * KSTR];        // V tile transposed, row = d, col = kv
    __shared__ __bf16 sP [4 * 16 * KSTR];    // per-wave P scratch (C->A relayout)
#if USE_TDM
    __shared__ float  sBias[2 * BM * BSTR];  // double-buffered TDM bias tile
#endif

    const int tid  = threadIdx.x;
    const int wave = tid >> 5;
    const int lane = tid & 31;
    const int lrow = lane & 15;
    const int half = lane >> 4;

    const int qblk  = blockIdx.x;            // 0..31
    const int bh    = blockIdx.y;            // 0..31
    const int qbase = qblk * BM + wave * 16; // this wave's 16 query rows

    const float*   Qb    = Q    + (size_t)bh * L_ * D_;
    const float*   Kb    = Kp   + (size_t)bh * L_ * D_;
    const float*   Vb    = Vp   + (size_t)bh * L_ * D_;
    const float*   BiasB = Bias + (size_t)bh * L_ * L_ + (size_t)(qblk * BM) * L_;
    const uint8_t* QMb   = QM + bh * L_;
    const uint8_t* VMb   = VM + bh * L_;

    // ---- Q A-fragments: 16 rows x 64 K in bf16, pre-scaled by 1/sqrt(D) ----
    // A operand (16-bit, 16x32), contiguous k order used here: lane = M, K pair = 16h+2v
    v16bf aQ[2];
    {
        const float* qr = Qb + (size_t)(qbase + lrow) * D_;
        #pragma unroll
        for (int c = 0; c < 2; ++c) {
            #pragma unroll
            for (int v = 0; v < 8; ++v) {
                int kk = 32 * c + 16 * half + 2 * v;
                aQ[c][2 * v]     = (__bf16)(qr[kk]     * SCALE);
                aQ[c][2 * v + 1] = (__bf16)(qr[kk + 1] * SCALE);
            }
        }
    }

    float mrow[8], lsum[8];
    v8f   accO[4];
    #pragma unroll
    for (int i = 0; i < 8; ++i) { mrow[i] = -3.0e38f; lsum[i] = 0.0f; }
    #pragma unroll
    for (int t = 0; t < 4; ++t) accO[t] = {};

    const int nkv = qblk + 1;   // causal: only kv blocks 0..qblk
    int buf = 0;

#if USE_TDM
    if (wave == 0)   // DMA the first bias tile while we stage K/V for block 0
        tdm_load_bias_tile(BiasB + 0, sBias + 0);
#endif

    for (int kvb = 0; kvb < nkv; ++kvb) {
        const int kvbase = kvb * BN;

        // ---- cooperative stage: K row-major bf16, V transposed bf16 ----
        {
            const int r  = tid >> 1;          // 0..63
            const int c0 = (tid & 1) * 32;    // 0 / 32
            const float4* krow = (const float4*)(Kb + (size_t)(kvbase + r) * D_ + c0);
            const float4* vrow = (const float4*)(Vb + (size_t)(kvbase + r) * D_ + c0);
            #pragma unroll
            for (int u = 0; u < 8; ++u) {
                float4 kf = krow[u];
                float4 vf = vrow[u];
                int c = c0 + 4 * u;
                sK[r * KSTR + c + 0] = (__bf16)kf.x;
                sK[r * KSTR + c + 1] = (__bf16)kf.y;
                sK[r * KSTR + c + 2] = (__bf16)kf.z;
                sK[r * KSTR + c + 3] = (__bf16)kf.w;
                sVT[(c + 0) * KSTR + r] = (__bf16)vf.x;
                sVT[(c + 1) * KSTR + r] = (__bf16)vf.y;
                sVT[(c + 2) * KSTR + r] = (__bf16)vf.z;
                sVT[(c + 3) * KSTR + r] = (__bf16)vf.w;
            }
            if (kvb + 1 < nkv) {  // pull next K/V tile toward L2/L0 early
                __builtin_prefetch(Kb + (size_t)(kvbase + BN + r) * D_ + c0, 0, 1);
                __builtin_prefetch(Vb + (size_t)(kvbase + BN + r) * D_ + c0, 0, 1);
            }
        }

#if USE_TDM
        if (wave == 0)
            __builtin_amdgcn_s_wait_tensorcnt(0);  // bias tile kvb landed in LDS
#endif
        __syncthreads();
#if USE_TDM
        if (wave == 0 && (kvb + 1) < nkv)          // overlap DMA of tile kvb+1
            tdm_load_bias_tile(BiasB + (kvbase + BN),
                               sBias + (buf ^ 1) * (BM * BSTR));
#endif

        // ---- S = (Q*scale) K^T + bias : bias preloaded into WMMA C ----
        v8f accS[4];
        const unsigned int* sK32 = (const unsigned int*)sK;
        #pragma unroll
        for (int t = 0; t < 4; ++t) {
#if USE_TDM
            const float* bp = sBias + buf * (BM * BSTR)
                            + (wave * 16 + 8 * half) * BSTR + (16 * t + lrow);
            #pragma unroll
            for (int i = 0; i < 8; ++i)
                accS[t][i] = bp[i * BSTR];
#else
            const float* bp = BiasB + (size_t)(wave * 16 + 8 * half) * L_
                                    + (kvbase + 16 * t + lrow);
            #pragma unroll
            for (int i = 0; i < 8; ++i)
                accS[t][i] = __builtin_nontemporal_load(bp + (size_t)i * L_);
#endif
            #pragma unroll
            for (int c = 0; c < 2; ++c) {
                // B operand (16-bit, 32x16), same contiguous k order as A: lane = N (kv col), K pair = 16h+2v
                v8u kb;
                #pragma unroll
                for (int v = 0; v < 8; ++v)
                    kb[v] = sK32[(16 * t + lrow) * (KSTR / 2)
                                 + (32 * c + 16 * half) / 2 + v];
                v16bf bK = __builtin_bit_cast(v16bf, kb);
                accS[t] = __builtin_amdgcn_wmma_f32_16x16x32_bf16(
                    false, aQ[c], false, bK, (short)0, accS[t], false, false);
            }
        }

        // ---- masking: causal AND values_mask -> NEG ----
        int colok[4];
        #pragma unroll
        for (int t = 0; t < 4; ++t)
            colok[t] = VMb[kvbase + 16 * t + lrow];
        #pragma unroll
        for (int t = 0; t < 4; ++t) {
            #pragma unroll
            for (int i = 0; i < 8; ++i) {
                int qr = qbase + i + 8 * half;
                int kc = kvbase + 16 * t + lrow;
                bool ok = (kc <= qr) && (colok[t] != 0);
                accS[t][i] = ok ? accS[t][i] : NEGV;
            }
        }

        // ---- online softmax (fp32): 16-wide lane reductions, wave32 ----
        #pragma unroll
        for (int i = 0; i < 8; ++i) {
            float mx = accS[0][i];
            #pragma unroll
            for (int t = 1; t < 4; ++t) mx = fmaxf(mx, accS[t][i]);
            #pragma unroll
            for (int off = 1; off < 16; off <<= 1)
                mx = fmaxf(mx, __shfl_xor(mx, off, 16));
            float mnew  = fmaxf(mrow[i], mx);
            float alpha = __expf(mrow[i] - mnew);
            mrow[i] = mnew;
            float rs = 0.0f;
            #pragma unroll
            for (int t = 0; t < 4; ++t) {
                float p = __expf(accS[t][i] - mnew);
                accS[t][i] = p;
                rs += p;
            }
            #pragma unroll
            for (int off = 1; off < 16; off <<= 1)
                rs += __shfl_xor(rs, off, 16);
            lsum[i] = lsum[i] * alpha + rs;
            #pragma unroll
            for (int t = 0; t < 4; ++t) accO[t][i] *= alpha;
        }

        // ---- P: C layout -> A layout via per-wave LDS (in-order DS, same wave) ----
        __bf16* sPw = sP + wave * 16 * KSTR;
        #pragma unroll
        for (int t = 0; t < 4; ++t)
            #pragma unroll
            for (int i = 0; i < 8; ++i)
                sPw[(i + 8 * half) * KSTR + 16 * t + lrow] = (__bf16)accS[t][i];

        const unsigned int* sP32 = (const unsigned int*)sPw;
        v16bf aP[2];
        #pragma unroll
        for (int c = 0; c < 2; ++c) {
            v8u pb;
            #pragma unroll
            for (int v = 0; v < 8; ++v) {
                int kk = 32 * c + 16 * half + 2 * v;
                pb[v] = sP32[lrow * (KSTR / 2) + kk / 2];
            }
            aP[c] = __builtin_bit_cast(v16bf, pb);
        }

        // ---- O += P V ----
        const unsigned int* sVT32 = (const unsigned int*)sVT;
        #pragma unroll
        for (int t = 0; t < 4; ++t) {         // t = output d-tile
            #pragma unroll
            for (int c = 0; c < 2; ++c) {     // c = kv chunk of 32
                v8u vb;
                #pragma unroll
                for (int v = 0; v < 8; ++v)
                    vb[v] = sVT32[(16 * t + lrow) * (KSTR / 2)
                                  + (32 * c + 16 * half) / 2 + v];
                v16bf bV = __builtin_bit_cast(v16bf, vb);
                accO[t] = __builtin_amdgcn_wmma_f32_16x16x32_bf16(
                    false, aP[c], false, bV, (short)0, accO[t], false, false);
            }
        }
        __syncthreads();
        buf ^= 1;
    }

    // ---- epilogue: normalize, apply queries_mask, non-temporal store ----
    float* ob = Out + (size_t)bh * L_ * D_;
    #pragma unroll
    for (int i = 0; i < 8; ++i) {
        int   qr  = qbase + i + 8 * half;
        float inv = (QMb[qr] != 0 && lsum[i] > 0.0f) ? (1.0f / lsum[i]) : 0.0f;
        #pragma unroll
        for (int t = 0; t < 4; ++t)
            __builtin_nontemporal_store(accO[t][i] * inv,
                                        ob + (size_t)qr * D_ + 16 * t + lrow);
    }
}

extern "C" void kernel_launch(void* const* d_in, const int* in_sizes, int n_in,
                              void* d_out, int out_size, void* d_ws, size_t ws_size,
                              hipStream_t stream) {
    (void)in_sizes; (void)n_in; (void)out_size; (void)d_ws; (void)ws_size;
    const float*   Q    = (const float*)d_in[0];
    const float*   K    = (const float*)d_in[1];
    const float*   V    = (const float*)d_in[2];
    const uint8_t* QM   = (const uint8_t*)d_in[3];   // jax bool -> 1 byte
    const uint8_t* VM   = (const uint8_t*)d_in[4];
    const float*   Bias = (const float*)d_in[5];
    float*         Out  = (float*)d_out;

    dim3 grid(L_ / BM, B_ * H_);   // 32 x 32 workgroups
    dim3 block(128);               // 4 wave32
    attn_bias_fwd_kernel<<<grid, block, 0, stream>>>(Q, K, V, QM, VM, Bias, Out);
}
